// APNB_23845658427902
// MI455X (gfx1250) — matrix operand order repaired and checked by reading, not compiled, not run on hardware
//
#include <hip/hip_runtime.h>

typedef __bf16 bf16;
typedef __attribute__((ext_vector_type(16))) __bf16 v16bf;
typedef __attribute__((ext_vector_type(8)))  float  v8f;
typedef unsigned int u32x4 __attribute__((ext_vector_type(4)));
typedef int i32x4 __attribute__((ext_vector_type(4)));
typedef int i32x8 __attribute__((ext_vector_type(8)));

union Frag   { v16bf v; uint4 u[2]; };
union B16x8  { uint4 u; bf16 h[8]; unsigned short s[8]; unsigned d[4]; };

#define HW   16384
#define CIN  512
#define CK   256
#define CO   512
#define NB   4
#define SPAD 112

static __device__ __forceinline__ v8f wmma_bf16(const Frag& a, const Frag& b, v8f c) {
  return __builtin_amdgcn_wmma_f32_16x16x32_bf16(false, a.v, false, b.v, (short)0, c, false, false);
}

// 32-bit LDS byte offset of a generic pointer that points into LDS.
static __device__ __forceinline__ unsigned lds_off(const void* p) {
  return (unsigned)(unsigned long long)(__attribute__((address_space(3))) const void*)p;
}

// ---- Tensor Data Mover: DMA a 2-D tile (tile_w elems x tile_h rows, 2-byte
// elements, row stride = stride elems) from global into LDS, packed row-major.
static __device__ __forceinline__ void tdm_load_2d(unsigned lds_byte, const void* gaddr,
                                                   unsigned tile_w, unsigned tile_h,
                                                   unsigned stride) {
  unsigned long long ga = (unsigned long long)gaddr;
  u32x4 g0;
  g0[0] = 1u;                                             // count=1 (valid), user mode
  g0[1] = lds_byte;                                       // lds_addr
  g0[2] = (unsigned)(ga & 0xffffffffu);                   // global_addr[31:0]
  g0[3] = (unsigned)((ga >> 32) & 0x01ffffffu) | 0x80000000u; // global_addr[56:32] | type=2
  i32x8 g1;
  g1[0] = (1 << 16);                                      // data_size=1 (2 bytes), mask=0
  g1[1] = (int)((stride & 0xffffu) << 16);                // tensor_dim0[15:0]
  g1[2] = (int)((stride >> 16) & 0xffffu) | (int)((tile_h & 0xffffu) << 16); // dim0 hi | dim1 lo
  g1[3] = (int)((tile_w & 0xffffu) << 16);                // dim1 hi=0 | tile_dim0
  g1[4] = (int)(tile_h & 0xffffu);                        // tile_dim1 | tile_dim2=0
  g1[5] = (int)stride;                                    // tensor_dim0_stride[31:0]
  g1[6] = 0;
  g1[7] = 0;
  i32x4 z4 = {0, 0, 0, 0};
#if defined(__clang_major__) && (__clang_major__ >= 23)
  i32x8 z8 = {0, 0, 0, 0, 0, 0, 0, 0};
  __builtin_amdgcn_tensor_load_to_lds(g0, g1, z4, z4, z8, 0);
#else
  __builtin_amdgcn_tensor_load_to_lds(g0, g1, z4, z4, 0);
#endif
}

// ---------------------------------------------------------------- fp32 -> bf16
__global__ __launch_bounds__(256) void cvt_bf16_kernel(const float* __restrict__ src,
                                                       bf16* __restrict__ dst, unsigned n8) {
  unsigned i = blockIdx.x * 256 + threadIdx.x;
  if (i >= n8) return;
  const float4* s = (const float4*)(src + (size_t)i * 8);
  float4 f0 = s[0], f1 = s[1];
  B16x8 cv;
  cv.h[0]=(bf16)f0.x; cv.h[1]=(bf16)f0.y; cv.h[2]=(bf16)f0.z; cv.h[3]=(bf16)f0.w;
  cv.h[4]=(bf16)f1.x; cv.h[5]=(bf16)f1.y; cv.h[6]=(bf16)f1.z; cv.h[7]=(bf16)f1.w;
  *(uint4*)(dst + (size_t)i * 8) = cv.u;
}

// ------------------------------------- combined [w_kq; w_v] -> bf16 [512][512]
__global__ __launch_bounds__(256) void cvt_wqv_kernel(const float* __restrict__ w_kq,
                                                      const float* __restrict__ w_v,
                                                      bf16* __restrict__ wqvb) {
  int idx = blockIdx.x * 256 + threadIdx.x;
  int o = idx >> 9, c = idx & 511;
  float v = (o < 256) ? w_kq[o * 512 + c] : w_v[(o - 256) * 512 + c];
  wqvb[idx] = (bf16)v;
}

// --------------------------------------------- W_eff = W_bot[:, :512] @ w_out
__global__ __launch_bounds__(256) void make_weff_kernel(const float* __restrict__ w_bot,
                                                        const float* __restrict__ w_out,
                                                        bf16* __restrict__ weffb) {
  int idx = blockIdx.x * 256 + threadIdx.x;
  int o = idx >> 8, c = idx & 255;
  float s = 0.f;
  for (int k = 0; k < 512; ++k) s += w_bot[o * 1024 + k] * w_out[k * 256 + c];
  weffb[idx] = (bf16)s;
}

// -------------------- b_eff = W_bot[:, :512] @ b_out + b_bot ; W_b2 cast bf16
__global__ __launch_bounds__(256) void make_beff_wb2_kernel(const float* __restrict__ w_bot,
                                                            const float* __restrict__ b_out,
                                                            const float* __restrict__ b_bot,
                                                            float* __restrict__ beff,
                                                            bf16* __restrict__ wb2b) {
  int idx = blockIdx.x * 256 + threadIdx.x;
  int o = idx >> 9, c = idx & 511;
  wb2b[idx] = (bf16)w_bot[o * 1024 + 512 + c];
  if (idx < 512) {
    float s = b_bot[idx];
    for (int k = 0; k < 512; ++k) s += w_bot[idx * 1024 + k] * b_out[k];
    beff[idx] = s;
  }
}

// ---------------------------------------------------------------- Q/V GEMM
// Double-buffered: TDM DMA of A tile + perm-packed B transpose for step k+1
// overlap the WMMA burst of step k; one barrier per K-step.
__global__ __launch_bounds__(256) void qv_gemm_kernel(const bf16* __restrict__ featsb,
                                                      const bf16* __restrict__ wqvb,
                                                      const float* __restrict__ b_kq,
                                                      const float* __restrict__ b_v,
                                                      bf16* __restrict__ qbuf,
                                                      bf16* __restrict__ vbuf) {
  __shared__ bf16 As[2][128][32];
  __shared__ bf16 Bs[2][128][32];   // transposed: [pixel][k]
  const int bx = blockIdx.x, by = blockIdx.y, b = blockIdx.z;
  const int t = threadIdx.x, lane = t & 31, wave = t >> 5;
  const int wm = wave >> 1, wn = wave & 1;          // 4x2 wave grid, 32x64 per wave
  const bool isQ = (by < 2);
  const float* bias = isQ ? (b_kq + by * 128) : (b_v + (by - 2) * 128);
  bf16* dst = (isQ ? qbuf : vbuf) + (size_t)b * CK * HW
            + (size_t)((isQ ? by : by - 2) * 128) * HW;
  const bf16* Abase = wqvb + (size_t)(by * 128) * CIN;
  const bf16* Bsrc  = featsb + (size_t)b * CIN * HW + bx * 128;

  v8f acc[2][4];
  #pragma unroll
  for (int i = 0; i < 2; ++i)
    #pragma unroll
    for (int j = 0; j < 4; ++j)
      #pragma unroll
      for (int k = 0; k < 8; ++k) acc[i][j][k] = 0.f;

  const int brow2 = (t >> 4) * 2;     // k-row pair 0,2,..,30
  const int bcol  = (t & 15) * 8;     // 8 pixels
  const int m = lane & 15, g = lane >> 4;
  const unsigned asOff[2] = { lds_off(&As[0][0][0]), lds_off(&As[1][0][0]) };

  auto stageB = [&](int kk, int buf) {
    const bf16* src = Bsrc + (size_t)(kk * 32 + brow2) * HW + bcol;
    B16x8 r0, r1;
    r0.u = *(const uint4*)src;
    r1.u = *(const uint4*)(src + HW);
    #pragma unroll
    for (int e = 0; e < 4; ++e) {       // v_perm_b32 byte-packing
      unsigned p0 = __builtin_amdgcn_perm(r1.d[e], r0.d[e], 0x05040100u);
      unsigned p1 = __builtin_amdgcn_perm(r1.d[e], r0.d[e], 0x07060302u);
      *(unsigned*)&Bs[buf][bcol + 2 * e][brow2]     = p0;
      *(unsigned*)&Bs[buf][bcol + 2 * e + 1][brow2] = p1;
    }
  };

  // prologue: fill buffer 0
  if (wave == 0) tdm_load_2d(asOff[0], Abase, 32u, 128u, (unsigned)CIN);
  stageB(0, 0);
  if (wave == 0) __builtin_amdgcn_s_wait_tensorcnt(0);
  __syncthreads();

  int buf = 0;
  for (int kk = 0; kk < CIN / 32; ++kk) {
    const int nxt = kk + 1;
    if (nxt < CIN / 32) {               // prefetch next step into other buffer
      if (wave == 0) tdm_load_2d(asOff[buf ^ 1], Abase + nxt * 32, 32u, 128u, (unsigned)CIN);
      stageB(nxt, buf ^ 1);
    }
    #pragma unroll
    for (int mt = 0; mt < 2; ++mt) {
      Frag a;
      a.u[0] = *(const uint4*)&As[buf][wm * 32 + mt * 16 + m][16 * g];
      a.u[1] = *(const uint4*)&As[buf][wm * 32 + mt * 16 + m][16 * g + 8];
      #pragma unroll
      for (int nt = 0; nt < 4; ++nt) {
        Frag bb;
        bb.u[0] = *(const uint4*)&Bs[buf][wn * 64 + nt * 16 + m][16 * g];
        bb.u[1] = *(const uint4*)&Bs[buf][wn * 64 + nt * 16 + m][16 * g + 8];
        acc[mt][nt] = wmma_bf16(a, bb, acc[mt][nt]);
      }
    }
    if (wave == 0 && nxt < CIN / 32) __builtin_amdgcn_s_wait_tensorcnt(0);
    __syncthreads();
    buf ^= 1;
  }
  #pragma unroll
  for (int mt = 0; mt < 2; ++mt)
    #pragma unroll
    for (int nt = 0; nt < 4; ++nt)
      #pragma unroll
      for (int r = 0; r < 8; ++r) {
        int row = wm * 32 + mt * 16 + r + 8 * g;
        float v = acc[mt][nt][r] + bias[row];
        if (isQ) v = fmaxf(v, 0.f);
        int col = bx * 128 + wn * 64 + nt * 16 + m;
        dst[(size_t)row * HW + col] = (bf16)v;
      }
}

// ---------------------------------------------------------------- PPM pooling
__global__ __launch_bounds__(128) void ppm_pool_kernel(const bf16* __restrict__ qbuf,
                                                       const bf16* __restrict__ vbuf,
                                                       bf16* __restrict__ keyT,
                                                       bf16* __restrict__ valP) {
  const int ck = blockIdx.x, b = blockIdx.y, which = blockIdx.z;
  const bf16* plane = (which == 0 ? qbuf : vbuf) + (size_t)b * CK * HW + (size_t)ck * HW;
  const int c = threadIdx.x;
  __shared__ float colsum[18][128];

  float a1 = 0.f, a3[3] = {0,0,0}, a6[6] = {0,0,0,0,0,0}, a8[8] = {0,0,0,0,0,0,0,0};
  for (int h = 0; h < 128; ++h) {
    float x = (float)plane[h * 128 + c];
    a1 += x;
    #pragma unroll
    for (int i = 0; i < 3; ++i) {
      int hs = (i * 128) / 3, he = ((i + 1) * 128 + 2) / 3;
      if (h >= hs && h < he) a3[i] += x;
    }
    #pragma unroll
    for (int i = 0; i < 6; ++i) {
      int hs = (i * 128) / 6, he = ((i + 1) * 128 + 5) / 6;
      if (h >= hs && h < he) a6[i] += x;
    }
    #pragma unroll
    for (int i = 0; i < 8; ++i) {
      int hs = (i * 128) / 8, he = ((i + 1) * 128 + 7) / 8;
      if (h >= hs && h < he) a8[i] += x;
    }
  }
  colsum[0][c] = a1;
  #pragma unroll
  for (int i = 0; i < 3; ++i) colsum[1 + i][c] = a3[i];
  #pragma unroll
  for (int i = 0; i < 6; ++i) colsum[4 + i][c] = a6[i];
  #pragma unroll
  for (int i = 0; i < 8; ++i) colsum[10 + i][c] = a8[i];
  __syncthreads();

  const int s = threadIdx.x;
  if (s < 110) {
    int scale, base, i, j;
    if (s < 1)       { scale = 1; base = 0;  i = 0;            j = 0; }
    else if (s < 10) { scale = 3; base = 1;  i = (s - 1) / 3;  j = (s - 1) % 3; }
    else if (s < 46) { scale = 6; base = 4;  i = (s - 10) / 6; j = (s - 10) % 6; }
    else             { scale = 8; base = 10; i = (s - 46) / 8; j = (s - 46) % 8; }
    int hs = (i * 128) / scale, he = ((i + 1) * 128 + scale - 1) / scale;
    int ws = (j * 128) / scale, we = ((j + 1) * 128 + scale - 1) / scale;
    float sum = 0.f;
    for (int cc = ws; cc < we; ++cc) sum += colsum[base + i][cc];
    float avg = sum / (float)((he - hs) * (we - ws));
    if (which == 0) keyT[(size_t)b * SPAD * CK + (size_t)s * CK + ck] = (bf16)avg;
    else            valP[(size_t)b * CK * 128 + (size_t)ck * 128 + s] = (bf16)avg;
  }
  if (which == 0) { if (s >= 110 && s < 112) keyT[(size_t)b * SPAD * CK + (size_t)s * CK + ck] = (bf16)0.f; }
  else            { if (s >= 110 && s < 128) valP[(size_t)b * CK * 128 + (size_t)ck * 128 + s] = (bf16)0.f; }
}

// ---------------------------------------------------------------- attention
__global__ __launch_bounds__(128) void attn_kernel(const bf16* __restrict__ qbuf,
                                                   const bf16* __restrict__ keyT,
                                                   const bf16* __restrict__ valP,
                                                   bf16* __restrict__ ctxbuf) {
  __shared__ bf16 qT[64][256];          // [px][ck]
  __shared__ bf16 attnS[4][16][128];    // per-wave [px][s padded to 128]
  const int b = blockIdx.y, px0 = blockIdx.x * 64;
  const int t = threadIdx.x, lane = t & 31, wave = t >> 5;

  { // stage q tile [256 ck][64 px] -> qT[px][ck]; ck-pair perm packing
    const int c0 = (t & 7) * 8;
    #pragma unroll
    for (int i = 0; i < 8; ++i) {
      int p = (t >> 3) + 16 * i;       // ck pair index 0..127
      const bf16* srcp = qbuf + (size_t)b * CK * HW + (size_t)(2 * p) * HW + px0 + c0;
      B16x8 r0, r1;
      r0.u = *(const uint4*)srcp;
      r1.u = *(const uint4*)(srcp + HW);
      #pragma unroll
      for (int e = 0; e < 4; ++e) {
        unsigned p0 = __builtin_amdgcn_perm(r1.d[e], r0.d[e], 0x05040100u);
        unsigned p1 = __builtin_amdgcn_perm(r1.d[e], r0.d[e], 0x07060302u);
        *(unsigned*)&qT[c0 + 2 * e][2 * p]     = p0;
        *(unsigned*)&qT[c0 + 2 * e + 1][2 * p] = p1;
      }
    }
  }
  __syncthreads();

  const int m = lane & 15, g = lane >> 4;
  const int pxw = wave * 16;

  // sim: [16 px] x [112 s], K = 256
  v8f simacc[7];
  #pragma unroll
  for (int nt = 0; nt < 7; ++nt)
    #pragma unroll
    for (int k = 0; k < 8; ++k) simacc[nt][k] = 0.f;

  const bf16* keyB = keyT + (size_t)b * SPAD * CK;
  for (int kk = 0; kk < 8; ++kk) {
    Frag a;
    a.u[0] = *(const uint4*)&qT[pxw + m][kk * 32 + 16 * g];
    a.u[1] = *(const uint4*)&qT[pxw + m][kk * 32 + 16 * g + 8];
    #pragma unroll
    for (int nt = 0; nt < 7; ++nt) {
      Frag bb;
      const bf16* kp = keyB + (size_t)(nt * 16 + m) * CK + kk * 32 + 16 * g;
      bb.u[0] = ((const uint4*)kp)[0];
      bb.u[1] = ((const uint4*)kp)[1];
      simacc[nt] = wmma_bf16(a, bb, simacc[nt]);
    }
  }

  // softmax over s (row = r + 8*g, s = nt*16 + m)
  const float scl = 0.0625f;             // 256^-0.5
  #pragma unroll
  for (int r = 0; r < 8; ++r) {
    float pv[7];
    float mx = -3.0e38f;
    #pragma unroll
    for (int nt = 0; nt < 7; ++nt) {
      int sidx = nt * 16 + m;
      float vv = (sidx < 110) ? simacc[nt][r] * scl : -__builtin_inff();
      pv[nt] = vv;
      mx = fmaxf(mx, vv);
    }
    #pragma unroll
    for (int o = 1; o < 16; o <<= 1) mx = fmaxf(mx, __shfl_xor(mx, o));
    float sum = 0.f;
    #pragma unroll
    for (int nt = 0; nt < 7; ++nt) { float p = __expf(pv[nt] - mx); pv[nt] = p; sum += p; }
    #pragma unroll
    for (int o = 1; o < 16; o <<= 1) sum += __shfl_xor(sum, o);
    float inv = 1.f / sum;
    int row = r + 8 * g;
    #pragma unroll
    for (int nt = 0; nt < 7; ++nt) attnS[wave][row][nt * 16 + m] = (bf16)(pv[nt] * inv);
    attnS[wave][row][112 + m] = (bf16)0.f;   // pad K 112..127
  }
  __syncthreads();

  // ctx^T = val[256 ck][128 s] x attn^T[128 s][16 px]  (coalesced stores)
  const bf16* valB = valP + (size_t)b * CK * 128;
  for (int mt = 0; mt < 16; ++mt) {
    v8f c;
    #pragma unroll
    for (int k = 0; k < 8; ++k) c[k] = 0.f;
    #pragma unroll
    for (int kk = 0; kk < 4; ++kk) {
      Frag a;
      const bf16* vp = valB + (size_t)(mt * 16 + m) * 128 + kk * 32;
      a.u[0] = *(const uint4*)(vp + 16 * g);
      a.u[1] = *(const uint4*)(vp + 16 * g + 8);
      Frag bb;
      bb.u[0] = *(const uint4*)&attnS[wave][m][kk * 32 + 16 * g];
      bb.u[1] = *(const uint4*)&attnS[wave][m][kk * 32 + 16 * g + 8];
      c = wmma_bf16(a, bb, c);
    }
    #pragma unroll
    for (int r = 0; r < 8; ++r) {
      int ck = mt * 16 + r + 8 * g;
      ctxbuf[(size_t)b * CK * HW + (size_t)ck * HW + px0 + pxw + m] = (bf16)c[r];
    }
  }
}

// ---------------------------------------------------------------- final GEMM
// out = relu(W_eff x ctx + W_b2 x feats + b_eff); 24 pipelined K-steps.
__global__ __launch_bounds__(256) void out_gemm_kernel(const bf16* __restrict__ weffb,
                                                       const bf16* __restrict__ wb2b,
                                                       const float* __restrict__ beff,
                                                       const bf16* __restrict__ ctxbuf,
                                                       const bf16* __restrict__ featsb,
                                                       float* __restrict__ out) {
  __shared__ bf16 As[2][128][32];
  __shared__ bf16 Bs[2][128][32];
  const int bx = blockIdx.x, by = blockIdx.y, b = blockIdx.z;
  const int t = threadIdx.x, lane = t & 31, wave = t >> 5;
  const int wm = wave >> 1, wn = wave & 1;
  const int NSTEP = 8 + 16;            // K=256 over ctx, then K=512 over feats

  v8f acc[2][4];
  #pragma unroll
  for (int i = 0; i < 2; ++i)
    #pragma unroll
    for (int j = 0; j < 4; ++j)
      #pragma unroll
      for (int k = 0; k < 8; ++k) acc[i][j][k] = 0.f;

  const int brow2 = (t >> 4) * 2;
  const int bcol  = (t & 15) * 8;
  const int m = lane & 15, g = lane >> 4;
  const unsigned asOff[2] = { lds_off(&As[0][0][0]), lds_off(&As[1][0][0]) };
  const bf16* A0 = weffb + (size_t)(by * 128) * CK;
  const bf16* A1 = wb2b  + (size_t)(by * 128) * CIN;
  const bf16* B0 = ctxbuf + (size_t)b * CK * HW + bx * 128;
  const bf16* B1 = featsb + (size_t)b * CIN * HW + bx * 128;

  auto aPtr = [&](int s) { return (s < 8) ? (A0 + s * 32) : (A1 + (s - 8) * 32); };
  auto bPtr = [&](int s) {
    return (s < 8) ? (B0 + (size_t)(s * 32) * HW) : (B1 + (size_t)((s - 8) * 32) * HW);
  };
  auto aStride = [&](int s) { return (s < 8) ? (unsigned)CK : (unsigned)CIN; };

  auto stageB = [&](const bf16* bsrc, int buf) {
    const bf16* src = bsrc + (size_t)brow2 * HW + bcol;
    B16x8 r0, r1;
    r0.u = *(const uint4*)src;
    r1.u = *(const uint4*)(src + HW);
    #pragma unroll
    for (int e = 0; e < 4; ++e) {
      unsigned p0 = __builtin_amdgcn_perm(r1.d[e], r0.d[e], 0x05040100u);
      unsigned p1 = __builtin_amdgcn_perm(r1.d[e], r0.d[e], 0x07060302u);
      *(unsigned*)&Bs[buf][bcol + 2 * e][brow2]     = p0;
      *(unsigned*)&Bs[buf][bcol + 2 * e + 1][brow2] = p1;
    }
  };

  if (wave == 0) tdm_load_2d(asOff[0], aPtr(0), 32u, 128u, aStride(0));
  stageB(bPtr(0), 0);
  if (wave == 0) __builtin_amdgcn_s_wait_tensorcnt(0);
  __syncthreads();

  int buf = 0;
  for (int s = 0; s < NSTEP; ++s) {
    const int nxt = s + 1;
    if (nxt < NSTEP) {
      if (wave == 0) tdm_load_2d(asOff[buf ^ 1], aPtr(nxt), 32u, 128u, aStride(nxt));
      stageB(bPtr(nxt), buf ^ 1);
    }
    #pragma unroll
    for (int mt = 0; mt < 2; ++mt) {
      Frag a;
      a.u[0] = *(const uint4*)&As[buf][wm * 32 + mt * 16 + m][16 * g];
      a.u[1] = *(const uint4*)&As[buf][wm * 32 + mt * 16 + m][16 * g + 8];
      #pragma unroll
      for (int nt = 0; nt < 4; ++nt) {
        Frag bb;
        bb.u[0] = *(const uint4*)&Bs[buf][wn * 64 + nt * 16 + m][16 * g];
        bb.u[1] = *(const uint4*)&Bs[buf][wn * 64 + nt * 16 + m][16 * g + 8];
        acc[mt][nt] = wmma_bf16(a, bb, acc[mt][nt]);
      }
    }
    if (wave == 0 && nxt < NSTEP) __builtin_amdgcn_s_wait_tensorcnt(0);
    __syncthreads();
    buf ^= 1;
  }
  #pragma unroll
  for (int mt = 0; mt < 2; ++mt)
    #pragma unroll
    for (int nt = 0; nt < 4; ++nt)
      #pragma unroll
      for (int r = 0; r < 8; ++r) {
        int row = by * 128 + wm * 32 + mt * 16 + r + 8 * g;
        float v = fmaxf(acc[mt][nt][r] + beff[row], 0.f);
        int col = bx * 128 + wn * 64 + nt * 16 + m;
        out[(size_t)b * CO * HW + (size_t)row * HW + col] = v;
      }
}

// ---------------------------------------------------------------- launcher
extern "C" void kernel_launch(void* const* d_in, const int* in_sizes, int n_in,
                              void* d_out, int out_size, void* d_ws, size_t ws_size,
                              hipStream_t stream) {
  (void)in_sizes; (void)n_in; (void)out_size; (void)ws_size;
  const float* feats = (const float*)d_in[0];
  const float* w_kq  = (const float*)d_in[1];
  const float* b_kq  = (const float*)d_in[2];
  const float* w_v   = (const float*)d_in[3];
  const float* b_v   = (const float*)d_in[4];
  const float* w_out = (const float*)d_in[5];
  const float* b_out = (const float*)d_in[6];
  const float* w_bot = (const float*)d_in[7];
  const float* b_bot = (const float*)d_in[8];
  float* out = (float*)d_out;

  char* ws = (char*)d_ws;
  size_t off = 0;
  auto alloc = [&](size_t bytes) {
    char* p = ws + off;
    off += (bytes + 255) & ~(size_t)255;
    return p;
  };
  bf16*  featsb = (bf16*)alloc((size_t)NB * CIN * HW * 2);   // 64 MiB
  bf16*  qbuf   = (bf16*)alloc((size_t)NB * CK * HW * 2);    // 32 MiB
  bf16*  vbuf   = (bf16*)alloc((size_t)NB * CK * HW * 2);    // 32 MiB
  bf16*  ctxbuf = (bf16*)alloc((size_t)NB * CK * HW * 2);    // 32 MiB
  bf16*  keyT   = (bf16*)alloc((size_t)NB * SPAD * CK * 2);
  bf16*  valP   = (bf16*)alloc((size_t)NB * CK * 128 * 2);
  bf16*  weffb  = (bf16*)alloc((size_t)CO * CK * 2);
  bf16*  wb2b   = (bf16*)alloc((size_t)CO * CIN * 2);
  bf16*  wqvb   = (bf16*)alloc((size_t)CIN * CIN * 2);
  float* beff   = (float*)alloc((size_t)CO * 4);

  const unsigned n8 = (unsigned)((size_t)NB * CIN * HW / 8);
  cvt_bf16_kernel<<<(n8 + 255) / 256, 256, 0, stream>>>(feats, featsb, n8);
  cvt_wqv_kernel<<<CIN * CIN / 256, 256, 0, stream>>>(w_kq, w_v, wqvb);
  make_weff_kernel<<<CO * CK / 256, 256, 0, stream>>>(w_bot, w_out, weffb);
  make_beff_wb2_kernel<<<CO * CIN / 256, 256, 0, stream>>>(w_bot, b_out, b_bot, beff, wb2b);
  qv_gemm_kernel<<<dim3(HW / 128, 4, NB), 256, 0, stream>>>(featsb, wqvb, b_kq, b_v, qbuf, vbuf);
  ppm_pool_kernel<<<dim3(CK, NB, 2), 128, 0, stream>>>(qbuf, vbuf, keyT, valP);
  attn_kernel<<<dim3(HW / 64, NB), 128, 0, stream>>>(qbuf, keyT, valP, ctxbuf);
  out_gemm_kernel<<<dim3(HW / 128, CO / 128, NB), 256, 0, stream>>>(weffb, wb2b, beff, ctxbuf, featsb, out);
}
